// PI2RNet_18348100289096
// MI455X (gfx1250) — matrix operand order repaired and checked by reading, not compiled, not run on hardware
//
#include <hip/hip_runtime.h>
#include <hip/hip_bf16.h>
#include <math.h>

typedef _Float16 v16h  __attribute__((ext_vector_type(16)));
typedef _Float16 half8 __attribute__((ext_vector_type(8)));
typedef float    v8f   __attribute__((ext_vector_type(8)));

#define BB   2
#define CC   32
#define HH   256
#define WW   256
#define OPSF 288          // psffilter channels
#define OTOT 320          // 288 psf + 18 offset + 9 mask + 5 pad
#define KTOT 288          // im2col depth: 9 taps * 32 ch
#define BLKW 64           // pixels per workgroup (row strip)
#define STRIP_HALFS (3*66*32)       // 3 rows x 66 cols x 32 ch
#define STRIP_BYTES (STRIP_HALFS*2) // 12672
#define PSF_LDS_BYTES (BLKW*OPSF*2) // 36864 (f16)

// ---------------------------------------------------------------- weights pack
// wpk[o][k], k = tap*32 + c, tap = ky*3+kx. Rows: [0,288)=f_conv, [288,306)=p_conv,
// [306,315)=m_conv, [315,320)=zero. bias packed the same way.
__global__ __launch_bounds__(256) void pack_weights_kernel(
    const float* __restrict__ pw, const float* __restrict__ pb,
    const float* __restrict__ fw, const float* __restrict__ fb,
    const float* __restrict__ mw, const float* __restrict__ mb,
    _Float16* __restrict__ wpk, float* __restrict__ bias)
{
  int idx = blockIdx.x * 256 + threadIdx.x;
  if (idx < OTOT * KTOT) {
    int o = idx / KTOT, k = idx - o * KTOT;
    int tap = k >> 5, c = k & 31;
    int ky = tap / 3, kx = tap - 3 * (tap / 3);
    float v = 0.f;
    if (o < 288)      v = fw[((o * 32 + c) * 3 + ky) * 3 + kx];
    else if (o < 306) v = pw[(((o - 288) * 32 + c) * 3 + ky) * 3 + kx];
    else if (o < 315) v = mw[(((o - 306) * 32 + c) * 3 + ky) * 3 + kx];
    wpk[idx] = (_Float16)v;
  }
  if (idx < OTOT) {
    float bv = 0.f;
    if (idx < 288)      bv = fb[idx];
    else if (idx < 306) bv = pb[idx - 288];
    else if (idx < 315) bv = mb[idx - 306];
    bias[idx] = bv;
  }
}

// ------------------------------------------------- cooperative LDS strip load
// s_in layout: half[(row*66 + col)*32 + c]; row 0..2 = image rows h-1..h+1,
// col 0..65 = image cols w0-1..w0+64, zero padded outside. f32 -> f16 convert.
__device__ __forceinline__ void load_strip(const float* __restrict__ src,
                                           int b, int h, int w0,
                                           _Float16* __restrict__ s_in)
{
  for (int idx = threadIdx.x; idx < STRIP_HALFS; idx += 256) {
    int c = idx / 198;
    int rem = idx - c * 198;
    int row = rem / 66, col = rem - row * 66;      // consecutive tid -> consecutive col (coalesced)
    int hs = h - 1 + row, ws = w0 - 1 + col;
    float v = 0.f;
    if (hs >= 0 && hs < HH && ws >= 0 && ws < WW)
      v = src[((b * CC + c) * HH + hs) * WW + ws];
    s_in[(row * 66 + col) * 32 + c] = (_Float16)v;
  }
}

// A fragment (16x32 f16): lane = 16g+m, K = t + 16g  => one contiguous
// 16-half LDS load at c-base 16g.
__device__ __forceinline__ v16h load_afrag(const _Float16* __restrict__ s_in,
                                           int pix, int s, int g)
{
  int ky = s / 3, kx = s - 3 * (s / 3);
  const _Float16* base = s_in + (((ky * 66) + pix + kx) << 5);
  return *(const v16h*)(base + 16 * g);
}

// B fragment (32x16 f16): lanes0-15 <-> N, K = t + 16g => one aligned 32B load.
__device__ __forceinline__ v16h load_bfrag(const _Float16* __restrict__ wpk,
                                           int orow, int s, int g)
{
  return *(const v16h*)(wpk + orow * KTOT + s * 32 + 16 * g);
}

// -------------------------------------------- pass A: offset + mask-logit GEMM
// Writes offs[pixel][32]: [0,18)=offset, [18,27)=mask logits, rest pad.
__global__ __launch_bounds__(256) void offmask_kernel(
    const float* __restrict__ fsize, const _Float16* __restrict__ wpk,
    const float* __restrict__ biasv, float* __restrict__ offs)
{
  extern __shared__ char smem[];
  _Float16* s_in = (_Float16*)smem;

  int bid = blockIdx.x;
  int w0 = (bid & 3) * BLKW;
  int h  = (bid >> 2) & 255;
  int b  = bid >> 10;

  load_strip(fsize, b, h, w0, s_in);
  __syncthreads();

  int lane = threadIdx.x & 31;
  int wv   = threadIdx.x >> 5;
  int n = lane & 15, g = lane >> 4;
  int mtile = wv >> 1, ntile = wv & 1;        // 4 M-tiles x 2 N-tiles = 8 waves
  int pix = mtile * 16 + (lane & 15);
  int obase = 288 + ntile * 16;

  v8f acc = {};
#pragma unroll
  for (int s = 0; s < 9; ++s) {
    v16h a  = load_afrag(s_in, pix, s, g);
    v16h bf = load_bfrag(wpk, obase + n, s, g);
    acc = __builtin_amdgcn_wmma_f32_16x16x32_f16(false, a, false, bf,
                                                 (short)0, acc, false, false);
  }
  float bv = biasv[obase + n];
  int pb0 = (b * HH + h) * WW + w0;
#pragma unroll
  for (int r = 0; r < 8; ++r) {
    int pl = mtile * 16 + r + 8 * g;
    offs[(size_t)(pb0 + pl) * 32 + ntile * 16 + n] = acc[r] + bv;
  }
}

__device__ __forceinline__ float sample_x(const float* __restrict__ x,
                                          int base, int qx, int qy)
{
  if (qx >= 1 && qx <= HH && qy >= 1 && qy <= WW)
    return x[base + (qx - 1) * WW + (qy - 1)];
  return 0.f;
}

// ------------------- pass B: PSF GEMM (WMMA -> LDS) fused with deform + FAC
__global__ __launch_bounds__(256) void fused_psf_kernel(
    const float* __restrict__ fctx, const float* __restrict__ x,
    const _Float16* __restrict__ wpk, const float* __restrict__ biasv,
    const float* __restrict__ offs, float* __restrict__ out)
{
  extern __shared__ char smem[];
  _Float16* s_in  = (_Float16*)smem;
  _Float16* s_psf = (_Float16*)(smem + STRIP_BYTES);

  int bid = blockIdx.x;
  int w0 = (bid & 3) * BLKW;
  int h  = (bid >> 2) & 255;
  int b  = bid >> 10;

  load_strip(fctx, b, h, w0, s_in);
  __syncthreads();

  // Phase 2: 64x288 GEMM tile, K=288, into LDS (f16). 72 wave-jobs over 8 waves.
  int lane = threadIdx.x & 31;
  int wv   = threadIdx.x >> 5;
  int n = lane & 15, g = lane >> 4;
  for (int job = wv; job < 72; job += 8) {
    int mtile = job & 3;
    int ntile = job >> 2;                     // 0..17 -> channels 0..287
    int pix = mtile * 16 + (lane & 15);
    v8f acc = {};
#pragma unroll
    for (int s = 0; s < 9; ++s) {
      v16h a  = load_afrag(s_in, pix, s, g);
      v16h bf = load_bfrag(wpk, ntile * 16 + n, s, g);
      acc = __builtin_amdgcn_wmma_f32_16x16x32_f16(false, a, false, bf,
                                                   (short)0, acc, false, false);
    }
    float bv = biasv[ntile * 16 + n];
#pragma unroll
    for (int r = 0; r < 8; ++r) {
      int pl = mtile * 16 + r + 8 * g;
      s_psf[pl * OPSF + ntile * 16 + n] = (_Float16)(acc[r] + bv);
    }
  }
  __syncthreads();

  // Phase 3: deformable sampling + modulation + FAC contraction.
  // 4 threads per pixel, 8 channels each.
  int p_local = threadIdx.x >> 2;
  int c0 = (threadIdx.x & 3) * 8;
  int w = w0 + p_local;

  float acc8[8];
#pragma unroll
  for (int c = 0; c < 8; ++c) acc8[c] = 0.f;

#pragma unroll
  for (int i = 0; i < 3; ++i) {
#pragma unroll
    for (int j = 0; j < 3; ++j) {
      int hs   = h + ((i == 0) ? -1 : 0);
      int wsrc = w + ((j == 0) ? -1 : 0);
      if (hs < 0 || wsrc < 0) continue;       // zero rows/cols from FAC pad
      int np = ((i + 2) % 3) * 3 + ((j + 2) % 3);
      const float* op = offs + (size_t)((b * HH + hs) * WW + wsrc) * 32;
      float px = op[np]     + (float)(np / 3 - 1) + (float)(hs + 1);
      float py = op[9 + np] + (float)(np % 3 - 1) + (float)(wsrc + 1);
      float mask = 1.0f / (1.0f + __expf(-op[18 + np]));
      float fx = floorf(px), fy = floorf(py);
      float pxc = fminf(fmaxf(px, 0.f), 257.f);
      float pyc = fminf(fmaxf(py, 0.f), 257.f);
      int qlx = (int)fminf(fmaxf(fx, 0.f), 257.f);
      int qly = (int)fminf(fmaxf(fy, 0.f), 257.f);
      int qrx = (int)fminf(fmaxf(fx + 1.f, 0.f), 257.f);
      int qry = (int)fminf(fmaxf(fy + 1.f, 0.f), 257.f);
      float glt = (1.f + ((float)qlx - pxc)) * (1.f + ((float)qly - pyc));
      float grb = (1.f - ((float)qrx - pxc)) * (1.f - ((float)qry - pyc));
      float glb = (1.f + ((float)qlx - pxc)) * (1.f - ((float)qry - pyc));
      float grt = (1.f - ((float)qrx - pxc)) * (1.f + ((float)qly - pyc));
      int tap = i * 3 + j;
#pragma unroll
      for (int c = 0; c < 8; ++c) {
        int cc = c0 + c;
        int base = (b * CC + cc) << 16;       // * H*W
        float xlt = sample_x(x, base, qlx, qly);
        float xrb = sample_x(x, base, qrx, qry);
        float xlb = sample_x(x, base, qlx, qry);
        float xrt = sample_x(x, base, qrx, qly);
        float val = glt * xlt + grb * xrb + glb * xlb + grt * xrt;
        float wc  = (float)s_psf[p_local * OPSF + cc * 9 + tap];
        acc8[c] += val * mask * wc;
      }
    }
  }
#pragma unroll
  for (int c = 0; c < 8; ++c) {
    int cc = c0 + c;
    out[((b * CC + cc) * HH + h) * WW + w] = acc8[c];
  }
}

extern "C" void kernel_launch(void* const* d_in, const int* in_sizes, int n_in,
                              void* d_out, int out_size, void* d_ws, size_t ws_size,
                              hipStream_t stream) {
  const float* f_size = (const float*)d_in[0];
  const float* f_ctx  = (const float*)d_in[1];
  const float* x      = (const float*)d_in[2];
  const float* pw     = (const float*)d_in[3];
  const float* pb     = (const float*)d_in[4];
  const float* fw     = (const float*)d_in[5];
  const float* fb     = (const float*)d_in[6];
  const float* mw     = (const float*)d_in[7];
  const float* mb     = (const float*)d_in[8];
  float* out = (float*)d_out;

  // workspace partition: wpk (f16 320x288) | bias (f32 320) | offs (f32 B*H*W*32)
  char* ws = (char*)d_ws;
  _Float16* wpk  = (_Float16*)ws;                       // 184320 B
  float*    bias = (float*)(ws + OTOT * KTOT * 2);      // 1280 B -> ends at 185600 (256-aligned)
  float*    offs = (float*)(ws + OTOT * KTOT * 2 + OTOT * 4); // 16 MB

  pack_weights_kernel<<<(OTOT * KTOT + 255) / 256, 256, 0, stream>>>(
      pw, pb, fw, fb, mw, mb, wpk, bias);

  int nblocks = BB * HH * (WW / BLKW);   // 2048
  offmask_kernel<<<nblocks, 256, STRIP_BYTES, stream>>>(f_size, wpk, bias, offs);
  fused_psf_kernel<<<nblocks, 256, STRIP_BYTES + PSF_LDS_BYTES, stream>>>(
      f_ctx, x, wpk, bias, offs, out);
}
